// DecomposableAttentionEntailment_80298708566351
// MI455X (gfx1250) — matrix operand order repaired and checked by reading, not compiled, not run on hardware
//
#include <hip/hip_runtime.h>

typedef __attribute__((ext_vector_type(16))) __bf16 v16bf;
typedef __attribute__((ext_vector_type(8)))  __bf16 v8bf;
typedef __attribute__((ext_vector_type(8)))  float  v8f;
typedef __attribute__((ext_vector_type(4)))  int    v4i;

// ---- CDNA5 async global->LDS copy (ASYNCcnt path), with portable fallback ----
#if __has_builtin(__builtin_amdgcn_global_load_async_to_lds_b128) && \
    __has_builtin(__builtin_amdgcn_s_wait_asynccnt)
#define USE_ASYNC_LDS 1
#else
#define USE_ASYNC_LDS 0
#endif

__device__ __forceinline__ void copy16_g2l(const __bf16* g, __bf16* l) {
#if USE_ASYNC_LDS
    __builtin_amdgcn_global_load_async_to_lds_b128(
        (__attribute__((address_space(1))) v4i*)(g),
        (__attribute__((address_space(3))) v4i*)(l), 0, 0);
#else
    *(uint4*)l = *(const uint4*)g;
#endif
}
__device__ __forceinline__ void async_join() {
#if USE_ASYNC_LDS
    __builtin_amdgcn_s_wait_asynccnt(0);
#endif
}

#define TM 128
#define TN 64
#define TK 32

// GEMM: C[b] = (relu?)( concat(A0,A1)[b] @ W[b] ),  everything bf16 in memory,
// f32 WMMA accumulation.  W is supplied TRANSPOSED: WT[n*ldw + k] == W[k][n],
// so both A and WT tiles are contiguous 16B chunks -> pure async B128 staging.
// Block: 128x64 tile, 256 thr = 8 waves, each wave = 1 m-tile x 4 n-tiles.
__global__ __launch_bounds__(256) void gemm_bf16_wmma(
    const __bf16* __restrict__ A0, const __bf16* __restrict__ A1, int K1,
    const __bf16* __restrict__ WT, int ldw,
    __bf16* __restrict__ C, int ldc, int K,
    long long batchA, long long batchW, long long batchC, int do_relu)
{
    __shared__ __align__(64) __bf16 aS[TM][TK];   // 8KB
    __shared__ __align__(64) __bf16 bS[TN][TK];   // 4KB

    const int tid = threadIdx.x, wave = tid >> 5, lane = tid & 31;
    const long long bz = blockIdx.z;
    const __bf16* A0b = A0 + bz * batchA;
    const __bf16* A1b = A1 + bz * batchA;
    const __bf16* Wb  = WT + bz * batchW;
    __bf16*       Cb  = C  + bz * batchC;

    const int m0 = blockIdx.x * TM;
    const int n0 = blockIdx.y * TN;
    const int K2 = K - K1;

    v8f acc[4] = {};

    const int arow = (wave << 4) + (lane & 15);
    const int kb8  = (lane >> 4) << 3;   // K-half base (0/8), unused: A operand uses kb16
    const int kb16 = (lane >> 4) << 4;   // B operand K-half base (0/16)

    for (int k0 = 0; k0 < K; k0 += TK) {
        // A tile: 128 rows x 4 x 16B chunks = 512 chunks, 2 per thread (async)
        #pragma unroll
        for (int i = 0; i < 2; ++i) {
            int idx = (i << 8) + tid;
            int r = idx >> 2, kg = (idx & 3) << 3;
            int gk = k0 + kg;
            const __bf16* src = (gk < K1)
                ? A0b + (long long)(m0 + r) * K1 + gk
                : A1b + (long long)(m0 + r) * K2 + (gk - K1);
            copy16_g2l(src, &aS[r][kg]);
        }
        // W tile: 64 rows x 4 chunks = 256 chunks, 1 per thread (async)
        {
            int r = tid >> 2, kg = (tid & 3) << 3;
            copy16_g2l(Wb + (long long)(n0 + r) * ldw + (k0 + kg), &bS[r][kg]);
        }
        async_join();
        __syncthreads();

        // A operand: contiguous K-half layout {kb16..kb16+15}, matching B
        v16bf av = *(const v16bf*)&aS[arow][kb16];

        #pragma unroll
        for (int t = 0; t < 4; ++t) {
            v16bf bv = *(const v16bf*)&bS[(t << 4) + (lane & 15)][kb16];
            acc[t] = __builtin_amdgcn_wmma_f32_16x16x32_bf16(false, av, false, bv,
                                                             (short)0, acc[t],
                                                             false, false);
        }
        __syncthreads();
    }

    const int crow = m0 + (wave << 4) + ((lane >> 4) << 3);
    #pragma unroll
    for (int t = 0; t < 4; ++t) {
        int ccol = n0 + (t << 4) + (lane & 15);
        #pragma unroll
        for (int v = 0; v < 8; ++v) {
            float x = acc[t][v];
            if (do_relu) x = fmaxf(x, 0.f);
            Cb[(long long)(crow + v) * ldc + ccol] = (__bf16)x;
        }
    }
}

// fp32 -> bf16 elementwise (n multiple of 4)
__global__ __launch_bounds__(256) void cvt_bf16(const float* __restrict__ in,
                                                __bf16* __restrict__ out, long long n)
{
    long long i = ((long long)blockIdx.x * 256 + threadIdx.x) * 4;
    if (i + 3 < n) {
        float4 v = *(const float4*)(in + i);
        out[i]     = (__bf16)v.x;
        out[i + 1] = (__bf16)v.y;
        out[i + 2] = (__bf16)v.z;
        out[i + 3] = (__bf16)v.w;
    }
}

// fp32 [R][Cn] -> bf16 transposed [Cn][R], 32x32 LDS tile, optional batch (z)
__global__ __launch_bounds__(256) void transpose_cvt(
    const float* __restrict__ in, __bf16* __restrict__ outT,
    int R, int Cn, long long bIn, long long bOut)
{
    __shared__ float t[32][33];
    const long long bz = blockIdx.z;
    const float* ib = in + bz * bIn;
    __bf16* ob = outT + bz * bOut;
    int r0 = blockIdx.x * 32, c0 = blockIdx.y * 32;
    int c = threadIdx.x & 31, rr = threadIdx.x >> 5;
    #pragma unroll
    for (int i = 0; i < 4; ++i) {
        int r = rr + (i << 3);
        t[r][c] = ib[(long long)(r0 + r) * Cn + (c0 + c)];
    }
    __syncthreads();
    #pragma unroll
    for (int i = 0; i < 4; ++i) {
        int crow = rr + (i << 3);
        ob[(long long)(c0 + crow) * R + (r0 + c)] = (__bf16)t[c][crow];
    }
}

// per-batch 128x128 softmax: rows -> attnP[p][q], cols -> attnHT[q][p] (bf16 io)
__global__ __launch_bounds__(128) void softmax_pq(
    const __bf16* __restrict__ S, __bf16* __restrict__ attnP, __bf16* __restrict__ attnHT)
{
    const long long base = (long long)blockIdx.x * 128 * 128;
    const int t = threadIdx.x;
    {
        const __bf16* row = S + base + (long long)t * 128;
        float m = -1e30f;
        for (int q = 0; q < 128; ++q) m = fmaxf(m, (float)row[q]);
        float s = 0.f;
        for (int q = 0; q < 128; ++q) s += __expf((float)row[q] - m);
        float inv = 1.f / s;
        __bf16* o = attnP + base + (long long)t * 128;
        for (int q = 0; q < 128; ++q) o[q] = (__bf16)(__expf((float)row[q] - m) * inv);
    }
    {
        const __bf16* col = S + base + t;
        float m = -1e30f;
        for (int p = 0; p < 128; ++p) m = fmaxf(m, (float)col[(long long)p * 128]);
        float s = 0.f;
        for (int p = 0; p < 128; ++p) s += __expf((float)col[(long long)p * 128] - m);
        float inv = 1.f / s;
        __bf16* o = attnHT + base + (long long)t * 128;
        for (int p = 0; p < 128; ++p)
            o[p] = (__bf16)(__expf((float)col[(long long)p * 128] - m) * inv);
    }
}

// out[b][h] = sum_{s<128} X[(b*128+s)][h]   (X: [B*128,512] bf16, f32 accum)
__global__ __launch_bounds__(256) void seq_sum(const __bf16* __restrict__ X,
                                               __bf16* __restrict__ out)
{
    int idx = blockIdx.x * 256 + threadIdx.x;
    int b = idx >> 9;
    int h = idx & 511;
    const __bf16* p = X + ((long long)b * 128) * 512 + h;
    float s = 0.f;
    #pragma unroll 4
    for (int i = 0; i < 128; ++i) s += (float)p[(long long)i * 512];
    out[(long long)b * 512 + h] = (__bf16)s;
}

// out[b] = softmax( hidden[b] @ scorer ),  hidden bf16, scorer/out fp32
__global__ __launch_bounds__(256) void score_softmax(
    const __bf16* __restrict__ hidden, const float* __restrict__ scorer,
    float* __restrict__ out, int B)
{
    int b = blockIdx.x * 256 + threadIdx.x;
    if (b >= B) return;
    const __bf16* h = hidden + (long long)b * 512;
    float s0 = 0.f, s1 = 0.f;
    for (int k = 0; k < 512; ++k) {
        float v = (float)h[k];
        s0 += v * scorer[2 * k];
        s1 += v * scorer[2 * k + 1];
    }
    float m = fmaxf(s0, s1);
    float e0 = __expf(s0 - m), e1 = __expf(s1 - m);
    float inv = 1.f / (e0 + e1);
    out[2 * b]     = e0 * inv;
    out[2 * b + 1] = e1 * inv;
}

extern "C" void kernel_launch(void* const* d_in, const int* in_sizes, int n_in,
                              void* d_out, int out_size, void* d_ws, size_t ws_size,
                              hipStream_t stream)
{
    const float* premise    = (const float*)d_in[0];
    const float* hypothesis = (const float*)d_in[1];
    const float* attend_w0  = (const float*)d_in[2];
    const float* attend_w1  = (const float*)d_in[3];
    const float* compare_w0 = (const float*)d_in[4];
    const float* compare_w1 = (const float*)d_in[5];
    const float* agg_w0     = (const float*)d_in[6];
    const float* agg_w1     = (const float*)d_in[7];
    const float* scorer     = (const float*)d_in[8];
    float* out = (float*)d_out;

    const int Bn = 512, P = 128, D = 512, H = 512;
    const int Mseq = Bn * P;                       // 65536
    const long long S = (long long)Mseq * D;       // 33.55M elems
    const long long T = (long long)Bn * P * P;     // 8.39M elems
    const long long BH = (long long)Bn * H;        // 262144

    __bf16* w = (__bf16*)d_ws;
    __bf16* bfP    = w;              // premise bf16 [B*128,512]
    __bf16* bfH    = w + S;
    __bf16* bfPT   = w + 2 * S;      // per-batch transposed [B][512][128]
    __bf16* bfHT   = w + 3 * S;
    __bf16* bufA   = w + 4 * S;      // proj_p -> c1
    __bf16* bufB   = w + 5 * S;      // proj_h -> c2
    __bf16* bufC   = w + 6 * S;      // ffn tmp -> alignP
    __bf16* bufD   = w + 7 * S;      // alignH
    __bf16* scoresB= w + 8 * S;      // [B,128,128]
    __bf16* attnP  = scoresB + T;
    __bf16* attnHT = attnP + T;
    __bf16* sums_p = attnHT + T;
    __bf16* sums_h = sums_p + BH;
    __bf16* agg1   = sums_h + BH;
    __bf16* agg2   = agg1 + BH;
    __bf16* aw0T   = agg2 + BH;                 // [512][512]
    __bf16* aw1T   = aw0T + 512LL * 512;
    __bf16* cw0T   = aw1T + 512LL * 512;        // [512][1024]
    __bf16* cw1T   = cw0T + 512LL * 1024;
    __bf16* gw0T   = cw1T + 512LL * 512;        // [512][1024]
    __bf16* gw1T   = gw0T + 512LL * 1024;

    dim3 blk(256);
    auto G = [](int M, int N, int Z) { return dim3(M / TM, N / TN, Z); };

    // --- one-time precision/layout conversion (bf16-resident pipeline) ---
    cvt_bf16<<<(unsigned)(S / 4 / 256), blk, 0, stream>>>(premise, bfP, S);
    cvt_bf16<<<(unsigned)(S / 4 / 256), blk, 0, stream>>>(hypothesis, bfH, S);
    transpose_cvt<<<dim3(4, 16, Bn), blk, 0, stream>>>(premise, bfPT, P, D,
                                                       (long long)P * D, (long long)P * D);
    transpose_cvt<<<dim3(4, 16, Bn), blk, 0, stream>>>(hypothesis, bfHT, P, D,
                                                       (long long)P * D, (long long)P * D);
    transpose_cvt<<<dim3(16, 16, 1), blk, 0, stream>>>(attend_w0, aw0T, 512, 512, 0, 0);
    transpose_cvt<<<dim3(16, 16, 1), blk, 0, stream>>>(attend_w1, aw1T, 512, 512, 0, 0);
    transpose_cvt<<<dim3(32, 16, 1), blk, 0, stream>>>(compare_w0, cw0T, 1024, 512, 0, 0);
    transpose_cvt<<<dim3(16, 16, 1), blk, 0, stream>>>(compare_w1, cw1T, 512, 512, 0, 0);
    transpose_cvt<<<dim3(32, 16, 1), blk, 0, stream>>>(agg_w0, gw0T, 1024, 512, 0, 0);
    transpose_cvt<<<dim3(16, 16, 1), blk, 0, stream>>>(agg_w1, gw1T, 512, 512, 0, 0);

    // --- Attend FFN ---
    gemm_bf16_wmma<<<G(Mseq, H, 1), blk, 0, stream>>>(bfP, bfP, D, aw0T, D,
                                                      bufC, H, D, 0, 0, 0, 1);
    gemm_bf16_wmma<<<G(Mseq, H, 1), blk, 0, stream>>>(bufC, bufC, H, aw1T, H,
                                                      bufA, H, H, 0, 0, 0, 1);
    gemm_bf16_wmma<<<G(Mseq, H, 1), blk, 0, stream>>>(bfH, bfH, D, aw0T, D,
                                                      bufC, H, D, 0, 0, 0, 1);
    gemm_bf16_wmma<<<G(Mseq, H, 1), blk, 0, stream>>>(bufC, bufC, H, aw1T, H,
                                                      bufB, H, H, 0, 0, 0, 1);

    // --- scores[b] = proj_p[b] @ proj_h[b]^T  (WT == row-major proj_h) ---
    gemm_bf16_wmma<<<G(P, P, Bn), blk, 0, stream>>>(bufA, bufA, H, bufB, H,
                                                    scoresB, P, H,
                                                    (long long)P * H, (long long)P * H,
                                                    (long long)P * P, 0);

    softmax_pq<<<Bn, 128, 0, stream>>>(scoresB, attnP, attnHT);

    // --- alignments: p2h = attnP @ hyp,  h2p = attnHT @ prem (WT = inputT) ---
    gemm_bf16_wmma<<<G(P, D, Bn), blk, 0, stream>>>(attnP, attnP, P, bfHT, P,
                                                    bufC, D, P,
                                                    (long long)P * P, (long long)P * D,
                                                    (long long)P * D, 0);
    gemm_bf16_wmma<<<G(P, D, Bn), blk, 0, stream>>>(attnHT, attnHT, P, bfPT, P,
                                                    bufD, D, P,
                                                    (long long)P * P, (long long)P * D,
                                                    (long long)P * D, 0);

    // --- Compare FFN (K-concat inside GEMM) + sequence sums ---
    gemm_bf16_wmma<<<G(Mseq, H, 1), blk, 0, stream>>>(bfP, bufC, D, cw0T, 2 * D,
                                                      bufA, H, 2 * D, 0, 0, 0, 1);
    gemm_bf16_wmma<<<G(Mseq, H, 1), blk, 0, stream>>>(bufA, bufA, H, cw1T, H,
                                                      bufB, H, H, 0, 0, 0, 1);
    seq_sum<<<(unsigned)(BH / 256), blk, 0, stream>>>(bufB, sums_p);

    gemm_bf16_wmma<<<G(Mseq, H, 1), blk, 0, stream>>>(bfH, bufD, D, cw0T, 2 * D,
                                                      bufA, H, 2 * D, 0, 0, 0, 1);
    gemm_bf16_wmma<<<G(Mseq, H, 1), blk, 0, stream>>>(bufA, bufA, H, cw1T, H,
                                                      bufB, H, H, 0, 0, 0, 1);
    seq_sum<<<(unsigned)(BH / 256), blk, 0, stream>>>(bufB, sums_h);

    // --- Aggregate FFN + scorer softmax ---
    gemm_bf16_wmma<<<G(Bn, H, 1), blk, 0, stream>>>(sums_p, sums_h, H, gw0T, 2 * H,
                                                    agg1, H, 2 * H, 0, 0, 0, 1);
    gemm_bf16_wmma<<<G(Bn, H, 1), blk, 0, stream>>>(agg1, agg1, H, gw1T, H,
                                                    agg2, H, H, 0, 0, 0, 1);
    score_softmax<<<2, blk, 0, stream>>>(agg2, scorer, out, Bn);
}
